// MoEModel_9783935500857
// MI455X (gfx1250) — matrix operand order repaired and checked by reading, not compiled, not run on hardware
//
#include <hip/hip_runtime.h>

typedef __bf16 bf16_t;
typedef __bf16 v16bf __attribute__((ext_vector_type(16)));
typedef __bf16 v8bf  __attribute__((ext_vector_type(8)));
typedef float  v8f   __attribute__((ext_vector_type(8)));

#define D_MODEL 1024
#define N_EXP   8

// ---------------------------------------------------------------------------
// Kernel 1: weight prep. fp32 row-major [e][k][n]  ->  bf16 column-major
// [e][n][k] so WMMA B-fragments (16 consecutive K of one column) are two
// contiguous 16B loads per lane. Also zeroes the routing counters.
// ---------------------------------------------------------------------------
__global__ void moe_prep(const float* __restrict__ W1, const float* __restrict__ W2,
                         bf16_t* __restrict__ w1cm, bf16_t* __restrict__ w2cm,
                         int* __restrict__ cnt)
{
  if (blockIdx.x == 0 && blockIdx.y == 0 && blockIdx.z == 0 && threadIdx.x < N_EXP)
    cnt[threadIdx.x] = 0;

  __shared__ float tile[32][33];
  const int mat = blockIdx.z >> 3;
  const int e   = blockIdx.z & 7;
  const float* src = (mat ? W2 : W1) + (size_t)e * D_MODEL * D_MODEL;
  bf16_t*      dst = (mat ? w2cm : w1cm) + (size_t)e * D_MODEL * D_MODEL;

  const int tx = threadIdx.x & 31;
  const int ty = threadIdx.x >> 5;          // 0..7
  const int kt = blockIdx.y * 32;
  const int nt = blockIdx.x * 32;

  #pragma unroll
  for (int i = 0; i < 32; i += 8)
    tile[ty + i][tx] = src[(size_t)(kt + ty + i) * D_MODEL + (nt + tx)];
  __syncthreads();
  #pragma unroll
  for (int i = 0; i < 32; i += 8)
    dst[(size_t)(nt + ty + i) * D_MODEL + (kt + tx)] = (bf16_t)tile[tx][ty + i];
}

// ---------------------------------------------------------------------------
// Kernel 2: gating. One wave32 per token. W_gate staged in LDS. Butterfly
// reduce -> every lane holds all 8 scores; redundant top-2 (tie = low index,
// matches lax.top_k). Lane 0 appends to per-expert token lists; the wave
// initializes out[tok] = w0*b2[e0] + w1*b2[e1] (bias term of the combine).
// ---------------------------------------------------------------------------
__global__ __launch_bounds__(256)
void moe_gate(const float* __restrict__ x, const float* __restrict__ Wg,
              const float* __restrict__ bg, const float* __restrict__ b2,
              float* __restrict__ out,
              int* __restrict__ cnt, int* __restrict__ list, float* __restrict__ wgt,
              int N)
{
  __shared__ float sWg[D_MODEL * N_EXP];   // 32 KB
  __shared__ float sbg[N_EXP];
  for (int i = threadIdx.x; i < D_MODEL * N_EXP; i += 256) sWg[i] = Wg[i];
  if (threadIdx.x < N_EXP) sbg[threadIdx.x] = bg[threadIdx.x];
  __syncthreads();

  const int wave = threadIdx.x >> 5;
  const int lane = threadIdx.x & 31;
  const int tok  = blockIdx.x * 8 + wave;
  if (tok >= N) return;

  const float* xrow = x + (size_t)tok * D_MODEL;
  float acc[N_EXP] = {};
  for (int i = 0; i < D_MODEL / 32; ++i) {
    const int k = i * 32 + lane;
    const float xv = xrow[k];
    const float4* wr = (const float4*)&sWg[k * N_EXP];
    float4 w0 = wr[0], w1 = wr[1];
    acc[0] += xv * w0.x; acc[1] += xv * w0.y; acc[2] += xv * w0.z; acc[3] += xv * w0.w;
    acc[4] += xv * w1.x; acc[5] += xv * w1.y; acc[6] += xv * w1.z; acc[7] += xv * w1.w;
  }
  #pragma unroll
  for (int e = 0; e < N_EXP; ++e) {
    #pragma unroll
    for (int off = 16; off; off >>= 1) acc[e] += __shfl_xor(acc[e], off, 32);
    acc[e] += sbg[e];
  }

  int e0 = 0; float v0 = acc[0];
  #pragma unroll
  for (int e = 1; e < N_EXP; ++e) if (acc[e] > v0) { v0 = acc[e]; e0 = e; }
  int e1 = (e0 == 0) ? 1 : 0; float v1 = acc[e1];
  #pragma unroll
  for (int e = 0; e < N_EXP; ++e)
    if (e != e0 && acc[e] > v1) { v1 = acc[e]; e1 = e; }

  if (lane == 0) {
    int s0 = atomicAdd(&cnt[e0], 1);
    list[e0 * N + s0] = tok;  wgt[e0 * N + s0] = v0;
    int s1 = atomicAdd(&cnt[e1], 1);
    list[e1 * N + s1] = tok;  wgt[e1 * N + s1] = v1;
  }

  const float* b2a = b2 + (size_t)e0 * D_MODEL;
  const float* b2b = b2 + (size_t)e1 * D_MODEL;
  float* orow = out + (size_t)tok * D_MODEL;
  for (int j = lane; j < D_MODEL; j += 32)
    orow[j] = v0 * b2a[j] + v1 * b2b[j];
}

// ---------------------------------------------------------------------------
// Kernel 3: expert MLP. One workgroup = (expert e, tile of 32 routed tokens),
// 8 waves; each wave owns BOTH 16-row M-tiles of one 128-column slab, so each
// B fragment is loaded once and feeds two WMMAs (2x arithmetic intensity vs
// an mi-split mapping). X tile staged in LDS as bf16, chunked-K [K/8][M][8].
// Phase 1: H = relu(X @ W1 + b1); H (bf16) overwrites the X buffer.
// Phase 2: Y = H @ W2; hardware f32 atomics combine w_tok * Y into out.
// ---------------------------------------------------------------------------
__global__ __launch_bounds__(256)
void moe_mlp(const float* __restrict__ x, const float* __restrict__ b1,
             const bf16_t* __restrict__ w1cm, const bf16_t* __restrict__ w2cm,
             const int* __restrict__ cnt, const int* __restrict__ list,
             const float* __restrict__ wgt, float* __restrict__ out, int N)
{
  const int e      = blockIdx.y;
  const int nslots = cnt[e];
  const int base   = blockIdx.x * 32;
  if (base >= nslots) return;

  __shared__ __align__(16) bf16_t Xb[32 * D_MODEL];   // 64 KB, chunked-K

  // ---- gather + fp32->bf16 convert: 256 threads = 32 rows x 8 segments ----
  {
    const int m   = threadIdx.x >> 3;      // 0..31 token row in tile
    const int seg = threadIdx.x & 7;       // 128 K-elements per segment
    const int slot = base + m;
    const int tok  = (slot < nslots) ? list[e * N + slot] : 0;
    const float* xr = x + (size_t)tok * D_MODEL;
    #pragma unroll
    for (int kk = 0; kk < 128; kk += 8) {
      const int k = seg * 128 + kk;
      float4 f0 = *(const float4*)(xr + k);
      float4 f1 = *(const float4*)(xr + k + 4);
      v8bf h;
      h[0] = (bf16_t)f0.x; h[1] = (bf16_t)f0.y; h[2] = (bf16_t)f0.z; h[3] = (bf16_t)f0.w;
      h[4] = (bf16_t)f1.x; h[5] = (bf16_t)f1.y; h[6] = (bf16_t)f1.z; h[7] = (bf16_t)f1.w;
      *(v8bf*)&Xb[(k >> 3) * 256 + m * 8] = h;
    }
  }
  __syncthreads();

  const int wid  = threadIdx.x >> 5;       // nj: 128-column slab, 0..7
  const int lane = threadIdx.x & 31;
  const int nj   = wid;
  const int hi   = lane >> 4;
  const int ln   = lane & 15;

  v8f acc[16] = {};                        // [mi*8 + nt]
  const bf16_t* w1e = w1cm + (size_t)e * D_MODEL * D_MODEL;
  const bf16_t* w2e = w2cm + (size_t)e * D_MODEL * D_MODEL;

  // ---------------- phase 1: H = relu(X @ W1 + b1) ----------------
  for (int k0 = 0; k0 < D_MODEL; k0 += 32) {
    const int cb = (k0 >> 3) + 2 * hi;
    v8bf a0lo = *(const v8bf*)&Xb[cb * 256 + ln * 8];
    v8bf a0hi = *(const v8bf*)&Xb[(cb + 1) * 256 + ln * 8];
    v8bf a1lo = *(const v8bf*)&Xb[cb * 256 + (16 + ln) * 8];
    v8bf a1hi = *(const v8bf*)&Xb[(cb + 1) * 256 + (16 + ln) * 8];
    v16bf a0 = __builtin_shufflevector(a0lo, a0hi, 0,1,2,3,4,5,6,7,8,9,10,11,12,13,14,15);
    v16bf a1 = __builtin_shufflevector(a1lo, a1hi, 0,1,2,3,4,5,6,7,8,9,10,11,12,13,14,15);
    #pragma unroll
    for (int nt = 0; nt < 8; ++nt) {
      const int n = nj * 128 + nt * 16 + ln;
      v16bf b = *(const v16bf*)(w1e + (size_t)n * D_MODEL + k0 + hi * 16);
      acc[nt]     = __builtin_amdgcn_wmma_f32_16x16x32_bf16(
          false, a0, false, b, (short)0, acc[nt],     false, false);
      acc[8 + nt] = __builtin_amdgcn_wmma_f32_16x16x32_bf16(
          false, a1, false, b, (short)0, acc[8 + nt], false, false);
    }
  }
  __syncthreads();   // all waves done reading X

  // bias + relu + bf16 H back into Xb (chunked-K layout, K == H columns)
  #pragma unroll
  for (int nt = 0; nt < 8; ++nt) {
    const int n = nj * 128 + nt * 16 + ln;
    const float bias = b1[e * D_MODEL + n];
    #pragma unroll
    for (int mi = 0; mi < 2; ++mi) {
      #pragma unroll
      for (int v = 0; v < 8; ++v) {
        float hv = acc[mi * 8 + nt][v] + bias;
        hv = hv > 0.0f ? hv : 0.0f;
        const int m = mi * 16 + v + 8 * hi;
        Xb[(n >> 3) * 256 + m * 8 + (n & 7)] = (bf16_t)hv;
      }
    }
  }
  __syncthreads();

  // ---------------- phase 2: Y = H @ W2 ----------------
  #pragma unroll
  for (int t = 0; t < 16; ++t) acc[t] = (v8f){};
  for (int k0 = 0; k0 < D_MODEL; k0 += 32) {
    const int cb = (k0 >> 3) + 2 * hi;
    v8bf a0lo = *(const v8bf*)&Xb[cb * 256 + ln * 8];
    v8bf a0hi = *(const v8bf*)&Xb[(cb + 1) * 256 + ln * 8];
    v8bf a1lo = *(const v8bf*)&Xb[cb * 256 + (16 + ln) * 8];
    v8bf a1hi = *(const v8bf*)&Xb[(cb + 1) * 256 + (16 + ln) * 8];
    v16bf a0 = __builtin_shufflevector(a0lo, a0hi, 0,1,2,3,4,5,6,7,8,9,10,11,12,13,14,15);
    v16bf a1 = __builtin_shufflevector(a1lo, a1hi, 0,1,2,3,4,5,6,7,8,9,10,11,12,13,14,15);
    #pragma unroll
    for (int nt = 0; nt < 8; ++nt) {
      const int n = nj * 128 + nt * 16 + ln;
      v16bf b = *(const v16bf*)(w2e + (size_t)n * D_MODEL + k0 + hi * 16);
      acc[nt]     = __builtin_amdgcn_wmma_f32_16x16x32_bf16(
          false, a0, false, b, (short)0, acc[nt],     false, false);
      acc[8 + nt] = __builtin_amdgcn_wmma_f32_16x16x32_bf16(
          false, a1, false, b, (short)0, acc[8 + nt], false, false);
    }
  }

  // ---------------- combine: out[tok] += w_tok * Y (HW f32 atomics) --------
  #pragma unroll
  for (int mi = 0; mi < 2; ++mi) {
    #pragma unroll
    for (int v = 0; v < 8; ++v) {
      const int m = mi * 16 + v + 8 * hi;
      const int slot = base + m;
      if (slot < nslots) {
        const int tok = list[e * N + slot];
        const float w = wgt[e * N + slot];
        float* orow = out + (size_t)tok * D_MODEL;
        #pragma unroll
        for (int nt = 0; nt < 8; ++nt) {
          const int n = nj * 128 + nt * 16 + ln;
          unsafeAtomicAdd(orow + n, acc[mi * 8 + nt][v] * w);
        }
      }
    }
  }
}

// ---------------------------------------------------------------------------
extern "C" void kernel_launch(void* const* d_in, const int* in_sizes, int n_in,
                              void* d_out, int out_size, void* d_ws, size_t ws_size,
                              hipStream_t stream) {
  const float* x   = (const float*)d_in[0];   // [N, 1024]
  const float* Wg  = (const float*)d_in[1];   // [1024, 8]
  const float* bg  = (const float*)d_in[2];   // [8]
  const float* W1  = (const float*)d_in[3];   // [8, 1024, 1024]
  const float* b1  = (const float*)d_in[4];   // [8, 1024]
  const float* W2  = (const float*)d_in[5];   // [8, 1024, 1024]
  const float* b2  = (const float*)d_in[6];   // [8, 1024]
  float* out = (float*)d_out;

  const int N = in_sizes[0] / D_MODEL;        // 8192 tokens

  // workspace layout
  char* ws = (char*)d_ws;
  int*    cnt  = (int*)ws;                                           // 8 ints
  int*    list = (int*)(ws + 256);                                   // [8][N]
  float*  wgt  = (float*)(ws + 256 + (size_t)N_EXP * N * 4);         // [8][N]
  size_t  woff = 256 + 2 * (size_t)N_EXP * N * 4;
  woff = (woff + 255) & ~(size_t)255;
  bf16_t* w1cm = (bf16_t*)(ws + woff);                               // 16 MB
  bf16_t* w2cm = w1cm + (size_t)N_EXP * D_MODEL * D_MODEL;           // 16 MB

  // 1) weight transpose/convert + counter reset
  dim3 gPrep(D_MODEL / 32, D_MODEL / 32, 2 * N_EXP);
  moe_prep<<<gPrep, 256, 0, stream>>>(W1, W2, w1cm, w2cm, cnt);

  // 2) gating / routing / bias-init of out
  moe_gate<<<N / 8, 256, 0, stream>>>(x, Wg, bg, b2, out, cnt, list, wgt, N);

  // 3) expert MLPs (expert-major, WMMA bf16)
  dim3 gMlp((N + 31) / 32, N_EXP);
  moe_mlp<<<gMlp, 256, 0, stream>>>(x, b1, w1cm, w2cm, cnt, list, wgt, out, N);
}
